// MultipleRoutersLayerWiseSteeringWhisperEncoder_74869869904297
// MI455X (gfx1250) — matrix operand order repaired and checked by reading, not compiled, not run on hardware
//
#include <hip/hip_runtime.h>
#include <stdint.h>

// ---------------------------------------------------------------------------
// Fused layer-wise-steering Whisper encoder pass for gfx1250 (MI455X).
//
// One workgroup (256 threads = 8 waves) owns a 16-token tile. The F=1280
// hidden dimension is split across the 8 waves: each wave keeps 10 WMMA
// C-tiles of h (16 tok x 160 f) in registers (stays below the VGPR-256 MSB
// boundary, several waves/SIMD occupancy).
// Per layer:
//   - each wave: partial logits^T = sum over its 5 K-chunks of
//     Wr^T-frag(A) x h^T-frag(B)            (v_wmma_f32_16x16x32_bf16)
//   - wave 0 reduces the 8 partials, does the 8-expert softmax, folds the
//     steering scale, packs gates to bf16 and broadcasts 4 dwords/token
//     via LDS (hi-lanes publish zeros) -> other waves rebuild the gate
//     B-fragment with a single ds_load_b128, no VALU masking
//   - each wave: h[cc] = sv^T-frag(A) x gate^T-frag(B) + h[cc]  (WMMA acc.)
//   - global_prefetch of next layer's weight fragments
// Epilogue: cross-wave layernorm reduction via LDS, f32 store.
// Weight fragments are pre-swizzled (fully padded, zeros materialized) into
// bf16 WMMA A-fragments in d_ws, so the hot loop does zero mask VALU work.
// ---------------------------------------------------------------------------

typedef float  v8f   __attribute__((ext_vector_type(8)));
typedef __bf16 v16bf __attribute__((ext_vector_type(16)));

#define LAYERS 32
#define FDIM   1280
#define NEXP   8
#define NTILE  80            // FDIM / 16  (combine C-tiles, total)
#define KCH    40            // FDIM / 32  (router K chunks, total)
#define WSPLIT 8             // waves per workgroup (F split)
#define TILES_PW (NTILE / WSPLIT)   // 10 tiles per wave
#define KCH_PW   (KCH / WSPLIT)     // 5 K-chunks per wave

// pack two f32 into one dword of bf16 (truncate): low16 = bf16(lo), hi16 = bf16(hi)
static __device__ __forceinline__ uint32_t pack_bf16(float lo, float hi_) {
  return __builtin_amdgcn_perm(__float_as_uint(hi_), __float_as_uint(lo), 0x07060302u);
}

static __device__ __forceinline__ v16bf frag_from(uint32_t d0, uint32_t d1, uint32_t d2, uint32_t d3,
                                                  uint32_t d4, uint32_t d5, uint32_t d6, uint32_t d7) {
  union { uint32_t u[8]; v16bf v; } x;
  x.u[0] = d0; x.u[1] = d1; x.u[2] = d2; x.u[3] = d3;
  x.u[4] = d4; x.u[5] = d5; x.u[6] = d6; x.u[7] = d7;
  return x.v;
}

// ---------------------------------------------------------------------------
// Pack router_w (L,F,E) f32 -> full 32-lane bf16 A-fragments of Wr^T
// (M=16 rows: experts 0..7 real, 8..15 zero; K=32).
// Storage: [L][KCH][32 lanes][8 dwords]. Lane = khalf*16 + mrow.
// K order used here: v_j K=16*khalf+2j,2j+1 (contiguous; B fragment uses the same).
// ---------------------------------------------------------------------------
__global__ void pack_router_kernel(const float* __restrict__ rw, uint32_t* __restrict__ dst) {
  int idx = blockIdx.x * blockDim.x + threadIdx.x;     // over L*KCH*32
  if (idx >= LAYERS * KCH * 32) return;
  const int ln    = idx & 31;
  const int kk    = (idx >> 5) % KCH;
  const int l     = idx / (KCH * 32);
  const int mrow  = ln & 15;          // matrix row (expert; >=8 padded)
  const int khalf = ln >> 4;
#pragma unroll
  for (int j = 0; j < 8; ++j) {
    const int K0 = 2 * j + 16 * khalf;
    const int f  = kk * 32 + K0;
    float a = 0.f, b = 0.f;
    if (mrow < NEXP) {
      a = rw[((size_t)l * FDIM + f) * NEXP + mrow];
      b = rw[((size_t)l * FDIM + f + 1) * NEXP + mrow];
    }
    dst[(size_t)idx * 8 + j] = pack_bf16(a, b);
  }
}

// ---------------------------------------------------------------------------
// Pack steering_vectors (L,E,F) f32 -> full 32-lane bf16 A-fragments of sv^T
// (M=16 f-rows, K=32: experts 0..7 real, K>=8 zero). Only v0..3 stored
// (v4..7 are all-zero for every lane and synthesized as constant regs).
// Storage: [L][NTILE][32 lanes][4 dwords]. v_j holds experts 2j, 2j+1
// (zero for lanes 16..31, which cover K=8..15).
// ---------------------------------------------------------------------------
__global__ void pack_sv_kernel(const float* __restrict__ sv, uint32_t* __restrict__ dst) {
  int idx = blockIdx.x * blockDim.x + threadIdx.x;     // over L*NTILE*32
  if (idx >= LAYERS * NTILE * 32) return;
  const int ln    = idx & 31;
  const int cc    = (idx >> 5) % NTILE;
  const int l     = idx / (NTILE * 32);
  const int mrow  = ln & 15;          // f-row within tile
  const int khalf = ln >> 4;
  const int f     = cc * 16 + mrow;
#pragma unroll
  for (int j = 0; j < 4; ++j) {
    float a = 0.f, b = 0.f;
    if (khalf == 0) {                 // K = 0..7 (real experts); khalf=1 -> K=8..15 = 0
      a = sv[((size_t)l * NEXP + 2 * j) * FDIM + f];
      b = sv[((size_t)l * NEXP + 2 * j + 1) * FDIM + f];
    }
    dst[(size_t)idx * 4 + j] = pack_bf16(a, b);
  }
}

// ---------------------------------------------------------------------------
// Main fused kernel. Block = 256 threads = 8 waves = one 16-token tile.
// ---------------------------------------------------------------------------
__global__ __launch_bounds__(256, 1) __attribute__((amdgpu_waves_per_eu(4)))
void steer_fused_kernel(
    const float* __restrict__ x,
    const float* __restrict__ router_b,
    const float* __restrict__ layer_scales,
    const float* __restrict__ ln_gamma,
    const float* __restrict__ ln_beta,
    const uint32_t* __restrict__ wfragW,
    const uint32_t* __restrict__ wfragS,
    float* __restrict__ out)
{
  __shared__ float red[2][WSPLIT][32][8];       // 16 KB partial-logits buffers
  __shared__ uint4 gbuf[2][32];                 //  1 KB packed-gate broadcast

  const int lane  = threadIdx.x & 31;
  const int w     = threadIdx.x >> 5;           // 0..7  (F-split wave id)
  const int tile  = blockIdx.x;                 // 0..1499
  const int t     = lane & 15;                  // token column (and A-row id)
  const int hi    = lane >> 4;
  const int token = tile * 16 + t;
  const int cbase = w * TILES_PW;               // this wave's global tile base
  const int kbase = w * KCH_PW;                 // this wave's K-chunk base

  const float* xrow = x + (size_t)token * FDIM + hi * 8;

  // h tiles (C/D layout): h[cl][r] = h(token t, f = (cbase+cl)*16 + hi*8 + r)
  v8f h[TILES_PW];
#pragma unroll
  for (int cl = 0; cl < TILES_PW; ++cl) {
    const int cc = cbase + cl;
    const float4 a = *(const float4*)(xrow + cc * 16);
    const float4 b = *(const float4*)(xrow + cc * 16 + 4);
    v8f hv;
    hv[0] = a.x; hv[1] = a.y; hv[2] = a.z; hv[3] = a.w;
    hv[4] = b.x; hv[5] = b.y; hv[6] = b.z; hv[7] = b.w;
    h[cl] = hv;
  }

  for (int l = 0; l < LAYERS; ++l) {
    const uint32_t* wl = wfragW + (size_t)l * KCH * 32 * 8;
    const uint32_t* sl = wfragS + (size_t)l * NTILE * 32 * 4;

    // Prefetch next layer's fragment stream for this wave (global_prefetch_b8).
    if (l + 1 < LAYERS) {
      const char* nw = (const char*)(wfragW + (size_t)(l + 1) * KCH * 32 * 8 +
                                     (size_t)kbase * 32 * 8);
      const char* ns = (const char*)(wfragS + (size_t)(l + 1) * NTILE * 32 * 4 +
                                     (size_t)cbase * 32 * 4);
      __builtin_prefetch(nw + lane * 160, 0, 3);   // covers 5120 B region
      __builtin_prefetch(ns + lane * 160, 0, 3);   // covers 5120 B region
    }

    // ------- partial router GEMM over this wave's K range -------
    v8f c = {};
#pragma unroll
    for (int kl = 0; kl < KCH_PW; ++kl) {
      const int kk = kbase + kl;
      // A fragment: fully padded in memory -> straight 32 B load, no masking
      const uint32_t* ap = wl + (size_t)(kk * 32 + lane) * 8;
      const uint4 wa = *(const uint4*)(ap);
      const uint4 wb = *(const uint4*)(ap + 4);
      const v16bf Aw = frag_from(wa.x, wa.y, wa.z, wa.w, wb.x, wb.y, wb.z, wb.w);

      // B fragment: h^T chunk (f = kk*32 .. kk*32+31) for this lane's token.
      // Pack local tiles to bf16, exchange the 4 packed dwords with lane^16.
      uint32_t packA[4], packB[4];
#pragma unroll
      for (int i = 0; i < 4; ++i) {
        packA[i] = pack_bf16(h[2 * kl][2 * i], h[2 * kl][2 * i + 1]);
        packB[i] = pack_bf16(h[2 * kl + 1][2 * i], h[2 * kl + 1][2 * i + 1]);
      }
      uint32_t bh[8];
#pragma unroll
      for (int i = 0; i < 4; ++i) {
        const uint32_t send = hi ? packA[i] : packB[i];
        const uint32_t recv = (uint32_t)__shfl_xor((int)send, 16, 32);
        bh[i]     = hi ? recv : packA[i];       // lo: K = 0..7 (tile 2kl); hi: K = 16..23 (tile 2kl+1)
        bh[4 + i] = hi ? packB[i] : recv;       // lo: K = 8..15 (tile 2kl); hi: K = 24..31 (tile 2kl+1)
      }
      const v16bf Bh = frag_from(bh[0], bh[1], bh[2], bh[3], bh[4], bh[5], bh[6], bh[7]);

      c = __builtin_amdgcn_wmma_f32_16x16x32_bf16(false, Aw, false, Bh, (short)0, c, false, false);
    }

    // ------- publish partial logits -------
    const int p = l & 1;
    float4 pa, pb;
    pa.x = c[0]; pa.y = c[1]; pa.z = c[2]; pa.w = c[3];
    pb.x = c[4]; pb.y = c[5]; pb.z = c[6]; pb.w = c[7];
    *(float4*)&red[p][w][lane][0] = pa;
    *(float4*)&red[p][w][lane][4] = pb;
    __syncthreads();

    // ------- wave 0: reduce partials, softmax, pack + broadcast gates -------
    if (w == 0) {
      float lg[8];
#pragma unroll
      for (int r = 0; r < 8; ++r) lg[r] = router_b[l * NEXP + r];
#pragma unroll
      for (int w2 = 0; w2 < WSPLIT; ++w2) {
        const float4 qa = *(const float4*)&red[p][w2][lane][0];
        const float4 qb = *(const float4*)&red[p][w2][lane][4];
        lg[0] += qa.x; lg[1] += qa.y; lg[2] += qa.z; lg[3] += qa.w;
        lg[4] += qb.x; lg[5] += qb.y; lg[6] += qb.z; lg[7] += qb.w;
      }
      float mx = lg[0];
#pragma unroll
      for (int r = 1; r < 8; ++r) mx = fmaxf(mx, lg[r]);
      float sum = 0.f;
#pragma unroll
      for (int r = 0; r < 8; ++r) { lg[r] = __expf(lg[r] - mx); sum += lg[r]; }
      const float scale = layer_scales[l] / sum;   // fold steering scale
#pragma unroll
      for (int r = 0; r < 8; ++r) lg[r] *= scale;
      uint4 q;
      q.x = hi ? 0u : pack_bf16(lg[0], lg[1]);     // hi-lanes publish zeros so
      q.y = hi ? 0u : pack_bf16(lg[2], lg[3]);     // readers need no masking
      q.z = hi ? 0u : pack_bf16(lg[4], lg[5]);
      q.w = hi ? 0u : pack_bf16(lg[6], lg[7]);
      gbuf[p][lane] = q;
    }
    __syncthreads();

    // ------- gate^T B-fragment: one ds_load_b128, zero VALU masking -------
    const uint4 g = gbuf[p][lane];
    const v16bf Bg = frag_from(g.x, g.y, g.z, g.w, 0u, 0u, 0u, 0u);

    // ------- combine: h[cl] += s * gate . sv  (WMMA accumulate) -------
#pragma unroll
    for (int cl = 0; cl < TILES_PW; ++cl) {
      const int cc = cbase + cl;
      const uint4 sw = *(const uint4*)(sl + (size_t)(cc * 32 + lane) * 4);
      const v16bf As = frag_from(sw.x, sw.y, sw.z, sw.w, 0u, 0u, 0u, 0u);
      h[cl] = __builtin_amdgcn_wmma_f32_16x16x32_bf16(false, As, false, Bg, (short)0, h[cl], false, false);
    }
  }

  // ---------------- layernorm over F per token (cross-wave) ----------------
  float s1 = 0.f, s2 = 0.f;
#pragma unroll
  for (int cl = 0; cl < TILES_PW; ++cl) {
#pragma unroll
    for (int j = 0; j < 8; ++j) {
      const float v = h[cl][j];
      s1 += v;
      s2 = fmaf(v, v, s2);
    }
  }
  float2* red2 = (float2*)&red[0][0][0][0];     // safe: last layer used red[1]
  red2[w * 32 + lane] = make_float2(s1, s2);
  __syncthreads();
  float S1 = 0.f, S2 = 0.f;
#pragma unroll
  for (int w2 = 0; w2 < WSPLIT; ++w2) {
    const float2 a = red2[w2 * 32 + t];
    const float2 b = red2[w2 * 32 + t + 16];
    S1 += a.x + b.x;
    S2 += a.y + b.y;
  }
  const float inv_n = 1.f / (float)FDIM;
  const float mean  = S1 * inv_n;
  const float var   = S2 * inv_n - mean * mean;
  const float rstd  = rsqrtf(var + 1e-5f);

  const float* grow = ln_gamma + hi * 8;
  const float* brow = ln_beta + hi * 8;
  float* orow = out + (size_t)token * FDIM + hi * 8;
#pragma unroll
  for (int cl = 0; cl < TILES_PW; ++cl) {
    const int cc = cbase + cl;
    const float4 ga = *(const float4*)(grow + cc * 16);
    const float4 gb = *(const float4*)(grow + cc * 16 + 4);
    const float4 ba = *(const float4*)(brow + cc * 16);
    const float4 bb = *(const float4*)(brow + cc * 16 + 4);
    float4 oa, ob;
    oa.x = (h[cl][0] - mean) * rstd * ga.x + ba.x;
    oa.y = (h[cl][1] - mean) * rstd * ga.y + ba.y;
    oa.z = (h[cl][2] - mean) * rstd * ga.z + ba.z;
    oa.w = (h[cl][3] - mean) * rstd * ga.w + ba.w;
    ob.x = (h[cl][4] - mean) * rstd * gb.x + bb.x;
    ob.y = (h[cl][5] - mean) * rstd * gb.y + bb.y;
    ob.z = (h[cl][6] - mean) * rstd * gb.z + bb.z;
    ob.w = (h[cl][7] - mean) * rstd * gb.w + bb.w;
    *(float4*)(orow + cc * 16) = oa;
    *(float4*)(orow + cc * 16 + 4) = ob;
  }
}

// ---------------------------------------------------------------------------
extern "C" void kernel_launch(void* const* d_in, const int* in_sizes, int n_in,
                              void* d_out, int out_size, void* d_ws, size_t ws_size,
                              hipStream_t stream) {
  (void)in_sizes; (void)n_in; (void)out_size; (void)ws_size;
  const float* x      = (const float*)d_in[0];   // [16,1500,1280]
  const float* sv     = (const float*)d_in[1];   // [32,8,1280]
  const float* rw     = (const float*)d_in[2];   // [32,1280,8]
  const float* rb     = (const float*)d_in[3];   // [32,8]
  const float* ls     = (const float*)d_in[4];   // [32]
  const float* gamma  = (const float*)d_in[5];   // [1280]
  const float* beta   = (const float*)d_in[6];   // [1280]

  uint32_t* wfragW = (uint32_t*)d_ws;                                           // 1,310,720 B
  uint32_t* wfragS = (uint32_t*)((char*)d_ws + (size_t)LAYERS * KCH * 32 * 32); // 1,310,720 B

  pack_router_kernel<<<dim3((LAYERS * KCH * 32 + 255) / 256), dim3(256), 0, stream>>>(rw, wfragW);
  pack_sv_kernel<<<dim3((LAYERS * NTILE * 32 + 255) / 256), dim3(256), 0, stream>>>(sv, wfragS);

  // 24000 tokens / 16 per workgroup -> 1500 workgroups of 8 waves.
  steer_fused_kernel<<<dim3(1500), dim3(256), 0, stream>>>(
      x, rb, ls, gamma, beta, wfragW, wfragS, (float*)d_out);
}
